// LSTMDecoder_73349451481724
// MI455X (gfx1250) — compile-verified, matrix operand order checked by reading, not run on hardware
//
#include <hip/hip_runtime.h>

// ---------------------------------------------------------------------------
// LSTM decoder (input_size = 0), H=1024, N=2048, LENGTH=48, out (N, H, L) f32.
// Strategy: batch rows are independent across the recurrence -> one block owns
// 32 rows for all 48 steps. h lives in double-buffered LDS (bf16), c lives in
// wave registers. Per step each wave computes 4 hidden-column tiles x 4 gates
// with v_wmma_f32_16x16x32_bf16 (K-loop over 1024). Weights converted once to
// bf16 in workspace (8 MB, resident in the 192 MB L2 across all 48 steps).
// ---------------------------------------------------------------------------

typedef __bf16 bf16_t;
typedef __bf16  v8bf  __attribute__((ext_vector_type(8)));
typedef __bf16  v16bf __attribute__((ext_vector_type(16)));
typedef float   v8f   __attribute__((ext_vector_type(8)));

#define HDIM     1024
#define GDIM     4096          // 4*H
#define SEQLEN   48
#define NBATCH   2048
#define M_TILE   32            // batch rows per block
#define THREADS  512           // 16 waves (wave32)
#define NWAVES   16
#define JT_WAVE  4             // hidden 16-col tiles per wave: 64 tiles / 16 waves
#define LDS_STRIDE 1032        // H + 8 bf16 pad -> 4-bank rotation per row

__device__ __forceinline__ float fsig(float x) {
  // sigmoid(x) = 1 / (1 + 2^(-x*log2 e))
  float e = __builtin_amdgcn_exp2f(-1.44269504f * x);
  return __builtin_amdgcn_rcpf(1.0f + e);
}
__device__ __forceinline__ float ftanh(float x) {
  // tanh(x) = (1 - 2^(-2x*log2 e)) / (1 + 2^(-2x*log2 e))
  float e = __builtin_amdgcn_exp2f(-2.88539008f * x);
  return (1.0f - e) * __builtin_amdgcn_rcpf(1.0f + e);
}

// --- prep: W_hh f32 -> bf16, bias = b_ih + b_hh --------------------------------
__global__ void lstm_prep(const float* __restrict__ W,
                          const float* __restrict__ bih,
                          const float* __restrict__ bhh,
                          bf16_t* __restrict__ Wb,
                          float* __restrict__ bias) {
  int i = blockIdx.x * blockDim.x + threadIdx.x;
  int stride = gridDim.x * blockDim.x;
  for (int idx = i; idx < GDIM * HDIM; idx += stride)
    Wb[idx] = (bf16_t)W[idx];
  if (i < GDIM) bias[i] = bih[i] + bhh[i];
}

// --- main persistent-per-rows LSTM kernel --------------------------------------
__global__ void __launch_bounds__(THREADS, 1)
lstm_kernel(const float* __restrict__ h_in,
            const bf16_t* __restrict__ Wb,
            const float* __restrict__ bias,
            float* __restrict__ out) {
  extern __shared__ bf16_t smem[];          // 2 buffers of M_TILE x LDS_STRIDE
  const int tid  = threadIdx.x;
  const int wave = tid >> 5;
  const int lane = tid & 31;
  const int hi16 = lane >> 4;               // lane half selects K sub-range
  const int ln   = lane & 15;               // row (A) / column (B,C,D) index
  const int r0   = blockIdx.x * M_TILE;     // first batch row of this block

  // Load initial h (f32 -> bf16) into buffer 0, coalesced.
  for (int f = tid; f < M_TILE * HDIM; f += THREADS) {
    int row = f >> 10, col = f & (HDIM - 1);
    smem[row * LDS_STRIDE + col] = (bf16_t)h_in[(size_t)(r0 + row) * HDIM + col];
  }
  __syncthreads();

  const v8f vzero = {0.f, 0.f, 0.f, 0.f, 0.f, 0.f, 0.f, 0.f};

  // c state: JT_WAVE col-tiles x 2 row-tiles x 8 accumulator elems per lane
  v8f creg[JT_WAVE][2];
#pragma unroll
  for (int jt = 0; jt < JT_WAVE; ++jt) {
    creg[jt][0] = vzero;
    creg[jt][1] = vzero;
  }

  // Per-lane bias scalars (column-dependent, invariant over steps).
  float br[JT_WAVE][4];
#pragma unroll
  for (int jt = 0; jt < JT_WAVE; ++jt) {
    const int j0 = (wave * JT_WAVE + jt) * 16;
#pragma unroll
    for (int g = 0; g < 4; ++g) br[jt][g] = bias[g * HDIM + j0 + ln];
  }

  int cur = 0;
  for (int t = 0; t < SEQLEN; ++t) {
    const bf16_t* hbuf = smem + cur * (M_TILE * LDS_STRIDE);
    bf16_t*       nbuf = smem + (cur ^ 1) * (M_TILE * LDS_STRIDE);

    for (int jt = 0; jt < JT_WAVE; ++jt) {
      const int j0 = (wave * JT_WAVE + jt) * 16;

      v8f acc[4][2];
#pragma unroll
      for (int g = 0; g < 4; ++g) { acc[g][0] = vzero; acc[g][1] = vzero; }

      // K loop: 1024 / 32 WMMA k-blocks
      for (int kb = 0; kb < HDIM / 32; ++kb) {
        const int k0 = kb * 32 + hi16 * 8;  // lanes 0-15: K 0-7/16-23; 16-31: 8-15/24-31

        // A fragments (h rows) from LDS, one per 16-row tile
        v16bf afrag[2];
#pragma unroll
        for (int mt = 0; mt < 2; ++mt) {
          const bf16_t* ap = hbuf + (mt * 16 + ln) * LDS_STRIDE + k0;
          v8bf lo = *(const v8bf*)(ap);
          v8bf hc = *(const v8bf*)(ap + 16);
          afrag[mt] = __builtin_shufflevector(lo, hc, 0,1,2,3,4,5,6,7,8,9,10,11,12,13,14,15);
        }

        // B fragments: W row (gate*H + j0 + ln), contiguous K -> shared by both M tiles
#pragma unroll
        for (int g = 0; g < 4; ++g) {
          const bf16_t* wp = Wb + ((size_t)(g * HDIM + j0 + ln)) * HDIM + k0;
          v8bf lo = *(const v8bf*)(wp);
          v8bf hc = *(const v8bf*)(wp + 16);
          v16bf bfrag = __builtin_shufflevector(lo, hc, 0,1,2,3,4,5,6,7,8,9,10,11,12,13,14,15);
          acc[g][0] = __builtin_amdgcn_wmma_f32_16x16x32_bf16(
              false, afrag[0], false, bfrag, (short)0, acc[g][0], false, false);
          acc[g][1] = __builtin_amdgcn_wmma_f32_16x16x32_bf16(
              false, afrag[1], false, bfrag, (short)0, acc[g][1], false, false);
        }
      }

      // Gate fusion + state update + writeback (to next h buffer and output)
#pragma unroll
      for (int mt = 0; mt < 2; ++mt) {
#pragma unroll
        for (int e = 0; e < 8; ++e) {
          float zi = acc[0][mt][e] + br[jt][0];
          float zf = acc[1][mt][e] + br[jt][1];
          float zg = acc[2][mt][e] + br[jt][2];
          float zo = acc[3][mt][e] + br[jt][3];
          float ig = fsig(zi);
          float fg = fsig(zf);
          float gg = ftanh(zg);
          float og = fsig(zo);
          float cn = fg * creg[jt][mt][e] + ig * gg;
          creg[jt][mt][e] = cn;
          float hn = og * ftanh(cn);

          const int m = mt * 16 + e + hi16 * 8;   // C/D layout: VGPR e -> M=e / e+8
          const int j = j0 + ln;
          nbuf[m * LDS_STRIDE + j] = (bf16_t)hn;
          out[((size_t)(r0 + m) * HDIM + j) * SEQLEN + t] = hn;
        }
      }
    }
    __syncthreads();   // next-step h buffer complete for all waves
    cur ^= 1;
  }
}

extern "C" void kernel_launch(void* const* d_in, const int* in_sizes, int n_in,
                              void* d_out, int out_size, void* d_ws, size_t ws_size,
                              hipStream_t stream) {
  const float* h_in = (const float*)d_in[0];
  const float* W    = (const float*)d_in[1];
  const float* bih  = (const float*)d_in[2];
  const float* bhh  = (const float*)d_in[3];
  float* out = (float*)d_out;

  // workspace: [0, 8MB) bf16 W, then 16 KB f32 bias
  bf16_t* Wb   = (bf16_t*)d_ws;
  float*  bias = (float*)((char*)d_ws + (size_t)GDIM * HDIM * sizeof(bf16_t));

  lstm_prep<<<1024, 256, 0, stream>>>(W, bih, bhh, Wb, bias);

  const size_t lds_bytes = 2u * M_TILE * LDS_STRIDE * sizeof(bf16_t); // ~129 KB of 320 KB
  lstm_kernel<<<NBATCH / M_TILE, THREADS, lds_bytes, stream>>>(h_in, Wb, bias, out);
}
